// Head_43456479101605
// MI455X (gfx1250) — matrix operand order repaired and checked by reading, not compiled, not run on hardware
//
#include <hip/hip_runtime.h>
#include <hip/hip_bf16.h>

// ---------------------------------------------------------------------------
// Causal single-head attention, MI455X (gfx1250, wave32, WMMA bf16).
//   x:[4,4096,2048] f32, Wk/Wq/Wv:[2048,128] f32  ->  out:[4,4096,128] f32
// Stage 1: q,k = x@W (bf16), v = x@Wv stored TRANSPOSED per 128-key block
//          (vt[b][jblk][h][t]) so stage 2 can stream it linearly.
// Stage 2: flash attention; QK^T and PV on v_wmma_f32_16x16x32_bf16,
//          V double-buffered in LDS via global_load_async_to_lds_b128
//          (ASYNCcnt), prefetched in the shadow of the QK^T WMMAs.
// ---------------------------------------------------------------------------

typedef __bf16 bf16_t;
typedef __attribute__((ext_vector_type(8)))  __bf16 v8bf;
typedef __attribute__((ext_vector_type(16))) __bf16 v16bf;
typedef __attribute__((ext_vector_type(8)))  float  v8f;

#define N_EMBD 2048
#define HEAD   128
#define TSEQ   4096
#define BATCH  4
#define BLK    128
#define PROJ_ROWS 64
#define NEG_INF (-3.0e38f)

static __device__ __forceinline__ v16bf pack2(v8bf a, v8bf b) {
  v16bf r;
#pragma unroll
  for (int i = 0; i < 8; ++i) { r[i] = a[i]; r[8 + i] = b[i]; }
  return r;
}

// Issue one 32KB block copy (global -> LDS) as 8 async b128 ops per wave.
// gsrc/LDS base must be 16B aligned; layout is a straight linear copy.
static __device__ __forceinline__ void async_vload(const bf16_t* gsrc,
                                                   unsigned lds_base,
                                                   int tid) {
#pragma unroll
  for (int e = 0; e < 8; ++e) {
    unsigned byteoff = (unsigned)(e * 256 + tid) * 16u;
    unsigned lds = lds_base + byteoff;
    unsigned long long ga =
        (unsigned long long)(const void*)gsrc + (unsigned long long)byteoff;
    asm volatile("global_load_async_to_lds_b128 %0, %1, off"
                 :: "v"(lds), "v"(ga)
                 : "memory");
  }
}

// ---------------------------------------------------------------------------
// Stage 1: fused q/k/v projection (x read exactly once from HBM).
// 256 threads = 8 waves; wave w owns tile-row (w&3) and n-half (w>>2).
// ---------------------------------------------------------------------------
__global__ __launch_bounds__(256) void proj_kernel(
    const float* __restrict__ x,
    const float* __restrict__ Wq, const float* __restrict__ Wk,
    const float* __restrict__ Wv,
    bf16_t* __restrict__ qb, bf16_t* __restrict__ kb,
    bf16_t* __restrict__ vtb)
{
  __shared__ __align__(32) bf16_t Wt[3][HEAD][32];  // W chunks, [sel][h][k]

  const int r0   = blockIdx.x * PROJ_ROWS;
  const int tid  = threadIdx.x;
  const int w    = tid >> 5;
  const int lane = tid & 31;
  const int hi   = lane >> 4;
  const int lo   = lane & 15;
  const int m    = w & 3;          // tile-row within 64-row block
  const int nh   = w >> 2;         // n-column half: cols [nh*4, nh*4+4)

  v8f acc[3][4] = {};
  const float* xrow = x + (size_t)(r0 + m * 16 + lo) * N_EMBD;

  for (int kc = 0; kc < N_EMBD; kc += 32) {
    for (int e = tid; e < 3 * 32 * HEAD; e += 256) {
      int sel = e >> 12;
      int rem = e & 4095;
      int k = rem >> 7, n = rem & 127;
      const float* W = (sel == 0) ? Wq : (sel == 1) ? Wk : Wv;
      Wt[sel][n][k] = (bf16_t)W[(size_t)(kc + k) * HEAD + n];
    }
    __syncthreads();

    // A fragment (shared by the three GEMMs), f32 -> bf16 in regs
    const float* p1 = xrow + kc + 16 * hi;
    const float* p2 = xrow + kc + 16 * hi + 8;
    v16bf afrag;
#pragma unroll
    for (int i = 0; i < 8; ++i) {
      afrag[i]     = (bf16_t)p1[i];
      afrag[8 + i] = (bf16_t)p2[i];
    }

#pragma unroll
    for (int sel = 0; sel < 3; ++sel) {
      v16bf bf[4];
#pragma unroll
      for (int nn = 0; nn < 4; ++nn)
        bf[nn] = *(const v16bf*)(&Wt[sel][(nh * 4 + nn) * 16 + lo][16 * hi]);
#pragma unroll
      for (int nn = 0; nn < 4; ++nn)
        acc[sel][nn] = __builtin_amdgcn_wmma_f32_16x16x32_bf16(
            false, afrag, false, bf[nn], (short)0, acc[sel][nn], false, false);
    }
    __syncthreads();
  }

  // q, k: row-major [row][h]
#pragma unroll
  for (int sel = 0; sel < 2; ++sel) {
    bf16_t* dst = (sel == 0) ? qb : kb;
#pragma unroll
    for (int nn = 0; nn < 4; ++nn)
#pragma unroll
      for (int r = 0; r < 8; ++r) {
        int row = r0 + m * 16 + r + 8 * hi;
        int col = (nh * 4 + nn) * 16 + lo;
        dst[(size_t)row * HEAD + col] = (bf16_t)acc[sel][nn][r];
      }
  }
  // v: transposed per 128-key block -> vt[((b*32 + jb)*128 + h)*128 + tl]
#pragma unroll
  for (int nn = 0; nn < 4; ++nn)
#pragma unroll
    for (int r = 0; r < 8; ++r) {
      int row = r0 + m * 16 + r + 8 * hi;     // flat b*T + t
      int bb  = row >> 12;
      int t   = row & (TSEQ - 1);
      int jb  = t >> 7;
      int tl  = t & 127;
      int col = (nh * 4 + nn) * 16 + lo;
      vtb[(((size_t)bb * 32 + jb) * HEAD + col) * BLK + tl] =
          (bf16_t)acc[2][nn][r];
    }
}

// ---------------------------------------------------------------------------
// Stage 2: flash attention, one (batch, 128-query) block per workgroup.
// ---------------------------------------------------------------------------
__global__ __launch_bounds__(256) void attn_kernel(
    const bf16_t* __restrict__ qb, const bf16_t* __restrict__ kb,
    const bf16_t* __restrict__ vtb, float* __restrict__ out)
{
  extern __shared__ __align__(32) char smem[];
  float*  Ss   = (float*) (smem);                 // [128 q][128 k] f32
                                                  // (P bf16 aliased per row)
  bf16_t* Vt0  = (bf16_t*)(smem + 65536);         // [128 h][128 t] buffer 0
  bf16_t* Vt1  = (bf16_t*)(smem + 98304);         // [128 h][128 t] buffer 1
  float*  Mrow = (float*) (smem + 131072);
  float*  Lrow = (float*) (smem + 131584);
  float*  Arow = (float*) (smem + 132096);

  const int b    = blockIdx.x >> 5;
  const int qi   = blockIdx.x & 31;
  const int tid  = threadIdx.x;
  const int w    = tid >> 5;
  const int lane = tid & 31;
  const int hi   = lane >> 4;
  const int lo   = lane & 15;
  const size_t boff = (size_t)b * TSEQ * HEAD;
  const bf16_t* vtbase = vtb + (size_t)b * 32 * HEAD * BLK;
  const unsigned ldsV0 = (unsigned)(uintptr_t)(void*)Vt0;
  const unsigned ldsV1 = (unsigned)(uintptr_t)(void*)Vt1;

  if (tid < BLK) { Mrow[tid] = NEG_INF; Lrow[tid] = 0.0f; }

  // Q fragments in registers: this wave's 16 query rows, all 128 H
  const bf16_t* qrow =
      qb + boff + (size_t)(qi * BLK + w * 16 + lo) * HEAD;
  v16bf qf[4];
#pragma unroll
  for (int c = 0; c < 4; ++c) {
    v8bf r1 = *(const v8bf*)(qrow + 32 * c + 16 * hi);
    v8bf r2 = *(const v8bf*)(qrow + 32 * c + 16 * hi + 8);
    qf[c] = pack2(r1, r2);
  }

  v8f acc[8] = {};
  const float scale = 0.022097086912079608f;  // 1/sqrt(N_EMBD)

  // kick off async copy of V block 0 into buffer 0
  async_vload(vtbase, ldsV0, tid);

  for (int j = 0; j <= qi; ++j) {
    const int cur = j & 1;
    __syncthreads();  // prev iter done with Ss(P) and the buffer we refill

    // prefetch next V block into the other buffer (runs under QK^T)
    if (j < qi)
      async_vload(vtbase + (size_t)(j + 1) * HEAD * BLK,
                  cur ? ldsV0 : ldsV1, tid);

    // S = scale * Q K^T (+ causal mask on diagonal block)
    const bf16_t* kblk = kb + boff + (size_t)(j * BLK) * HEAD;
#pragma unroll
    for (int n = 0; n < 8; ++n) {
      const bf16_t* krow = kblk + (size_t)(n * 16 + lo) * HEAD;
      v16bf kf[4];
#pragma unroll
      for (int c = 0; c < 4; ++c)
        kf[c] = *(const v16bf*)(krow + 32 * c + 16 * hi);
      v8f s = {};
#pragma unroll
      for (int c = 0; c < 4; ++c)
        s = __builtin_amdgcn_wmma_f32_16x16x32_bf16(
            false, qf[c], false, kf[c], (short)0, s, false, false);
#pragma unroll
      for (int r = 0; r < 8; ++r) {
        int qm = w * 16 + r + 8 * hi;
        int kn = n * 16 + lo;
        float sv = s[r] * scale;
        if (j == qi && kn > qm) sv = NEG_INF;
        Ss[qm * BLK + kn] = sv;
      }
    }
    __syncthreads();

    // online softmax: one thread per query row; P (bf16) overwrites the
    // front half of the same f32 row (read index stays ahead of write).
    if (tid < BLK) {
      float mold = Mrow[tid];
      float mnew = mold;
      const float* srow = Ss + tid * BLK;
#pragma unroll 8
      for (int kk = 0; kk < BLK; ++kk) mnew = fmaxf(mnew, srow[kk]);
      float alpha = __expf(mold - mnew);
      float sum = 0.0f;
      bf16_t* prow = (bf16_t*)(Ss + tid * BLK);
#pragma unroll 8
      for (int kk = 0; kk < BLK; ++kk) {
        float p = __expf(srow[kk] - mnew);
        sum += p;
        prow[kk] = (bf16_t)p;
      }
      Mrow[tid] = mnew;
      Lrow[tid] = alpha * Lrow[tid] + sum;
      Arow[tid] = alpha;
    }

    // current V buffer resident (prefetch, if any, may still be in flight)
    if (j < qi)
      asm volatile("s_wait_asynccnt 0x8" ::: "memory");
    else
      asm volatile("s_wait_asynccnt 0x0" ::: "memory");
    __syncthreads();

    // rescale running O and accumulate P @ V
    float al[8];
#pragma unroll
    for (int r = 0; r < 8; ++r) al[r] = Arow[w * 16 + r + 8 * hi];
#pragma unroll
    for (int n = 0; n < 8; ++n)
#pragma unroll
      for (int r = 0; r < 8; ++r) acc[n][r] *= al[r];

    const bf16_t* Vt = cur ? Vt1 : Vt0;
    const bf16_t* prow = (const bf16_t*)(Ss + (size_t)(w * 16 + lo) * BLK);
#pragma unroll
    for (int c = 0; c < 4; ++c) {
      v8bf r1 = *(const v8bf*)(prow + 32 * c + 16 * hi);
      v8bf r2 = *(const v8bf*)(prow + 32 * c + 16 * hi + 8);
      v16bf afrag = pack2(r1, r2);
#pragma unroll
      for (int g = 0; g < 2; ++g) {
        v16bf vf[4];
#pragma unroll
        for (int n = 0; n < 4; ++n)
          vf[n] = *(const v16bf*)(
              &Vt[((g * 4 + n) * 16 + lo) * BLK + 32 * c + 16 * hi]);
#pragma unroll
        for (int n = 0; n < 4; ++n)
          acc[g * 4 + n] = __builtin_amdgcn_wmma_f32_16x16x32_bf16(
              false, afrag, false, vf[n], (short)0, acc[g * 4 + n],
              false, false);
      }
    }
  }
  __syncthreads();

  // normalize and store f32 output
  float linv[8];
#pragma unroll
  for (int r = 0; r < 8; ++r) {
    float l = Lrow[w * 16 + r + 8 * hi];
    linv[r] = (l > 0.0f) ? (1.0f / l) : 0.0f;
  }
#pragma unroll
  for (int n = 0; n < 8; ++n)
#pragma unroll
    for (int r = 0; r < 8; ++r) {
      int row = qi * BLK + w * 16 + r + 8 * hi;
      int col = n * 16 + lo;
      out[boff + (size_t)row * HEAD + col] = acc[n][r] * linv[r];
    }
}

// ---------------------------------------------------------------------------
extern "C" void kernel_launch(void* const* d_in, const int* in_sizes, int n_in,
                              void* d_out, int out_size, void* d_ws,
                              size_t ws_size, hipStream_t stream) {
  (void)in_sizes; (void)n_in; (void)out_size; (void)ws_size;
  const float* x  = (const float*)d_in[0];
  const float* Wk = (const float*)d_in[1];
  const float* Wq = (const float*)d_in[2];
  const float* Wv = (const float*)d_in[3];
  float* out = (float*)d_out;

  const size_t elems = (size_t)BATCH * TSEQ * HEAD;
  bf16_t* qbuf  = (bf16_t*)d_ws;
  bf16_t* kbuf  = qbuf + elems;
  bf16_t* vtbuf = kbuf + elems;

  proj_kernel<<<dim3(BATCH * TSEQ / PROJ_ROWS), 256, 0, stream>>>(
      x, Wq, Wk, Wv, qbuf, kbuf, vtbuf);

  const size_t smem = 65536u + 2u * 32768u + 3u * 512u;  // 132608 B
  (void)hipFuncSetAttribute((const void*)attn_kernel,
                            hipFuncAttributeMaxDynamicSharedMemorySize,
                            (int)smem);
  attn_kernel<<<dim3(BATCH * TSEQ / BLK), 256, smem, stream>>>(
      qbuf, kbuf, vtbuf, out);
}
